// TransformerBlock_5978594476355
// MI455X (gfx1250) — compile-verified, matrix operand order checked by reading, not run on hardware
//
#include <hip/hip_runtime.h>
#include <hip/hip_bf16.h>

typedef __attribute__((ext_vector_type(16))) __bf16 v16bf;
typedef __attribute__((ext_vector_type(8)))  float  v8f;

// ---------------------------------------------------------------------------
// CDNA5 async global->LDS copy (ASYNCcnt-tracked DMA, no VGPR round trip).
// Generic pointers to LDS carry the LDS byte offset in their low 32 bits.
// ---------------------------------------------------------------------------
__device__ __forceinline__ void async_ld_b128(const void* g, void* l) {
    unsigned int loff = (unsigned int)(unsigned long long)l;
    asm volatile("global_load_async_to_lds_b128 %0, %1, off"
                 :: "v"(loff), "v"((unsigned long long)g) : "memory");
}
__device__ __forceinline__ void wait_async() {
    asm volatile("s_wait_asynccnt 0x0" ::: "memory");
}

// ---------------------------------------------------------------------------
// Reductions (wave32)
// ---------------------------------------------------------------------------
__device__ __forceinline__ float wave_sum(float v) {
    #pragma unroll
    for (int o = 16; o; o >>= 1) v += __shfl_xor(v, o, 32);
    return v;
}
__device__ __forceinline__ float wave_max(float v) {
    #pragma unroll
    for (int o = 16; o; o >>= 1) v = fmaxf(v, __shfl_xor(v, o, 32));
    return v;
}

// ---------------------------------------------------------------------------
// Tiled transpose (+ convert) -> bf16.  in: R x C (row-major), out: C x R.
// 32x32 LDS tile, 256 threads (32x8), optional batch via blockIdx.z.
// ---------------------------------------------------------------------------
template <typename TIn>
__global__ __launch_bounds__(256)
void transpose_to_bf16(const TIn* __restrict__ in, __bf16* __restrict__ out,
                       int R, int C, long inBatch, long outBatch) {
    __shared__ __bf16 t[32][33];
    in  += (long)blockIdx.z * inBatch;
    out += (long)blockIdx.z * outBatch;
    const int bx = blockIdx.x * 32;           // C-dim tile
    const int by = blockIdx.y * 32;           // R-dim tile
    const int tx = threadIdx.x & 31, ty = threadIdx.x >> 5;
    #pragma unroll
    for (int r = ty; r < 32; r += 8)
        t[r][tx] = (__bf16)in[(long)(by + r) * C + bx + tx];
    __syncthreads();
    #pragma unroll
    for (int r = ty; r < 32; r += 8)
        out[(long)(bx + r) * R + by + tx] = t[tx][r];
}

// ---------------------------------------------------------------------------
// LayerNorm: one block per token row (D = 1024, 256 threads x float4)
// ---------------------------------------------------------------------------
__global__ __launch_bounds__(256) void ln_bf16(const float* __restrict__ x,
                                               const float* __restrict__ g,
                                               const float* __restrict__ bb,
                                               __bf16* __restrict__ out, int Dd) {
    const int  row = blockIdx.x;
    const int  tid = threadIdx.x;
    const long base = (long)row * Dd;
    float4 xv = ((const float4*)(x + base))[tid];

    float s = xv.x + xv.y + xv.z + xv.w;
    float q = xv.x * xv.x + xv.y * xv.y + xv.z * xv.z + xv.w * xv.w;
    s = wave_sum(s); q = wave_sum(q);

    __shared__ float red[16];
    if ((tid & 31) == 0) { red[tid >> 5] = s; red[8 + (tid >> 5)] = q; }
    __syncthreads();
    float ts = 0.f, tq = 0.f;
    #pragma unroll
    for (int i = 0; i < 8; i++) { ts += red[i]; tq += red[8 + i]; }

    const float mu  = ts / Dd;
    const float var = tq / Dd - mu * mu;
    const float rs  = rsqrtf(var + 1e-5f);

    const int c = tid * 4;
    float4 gv = ((const float4*)g)[tid];
    float4 bv = ((const float4*)bb)[tid];
    out[base + c + 0] = (__bf16)((xv.x - mu) * rs * gv.x + bv.x);
    out[base + c + 1] = (__bf16)((xv.y - mu) * rs * gv.y + bv.y);
    out[base + c + 2] = (__bf16)((xv.z - mu) * rs * gv.z + bv.z);
    out[base + c + 3] = (__bf16)((xv.w - mu) * rs * gv.w + bv.w);
}

// ---------------------------------------------------------------------------
// Masked softmax over one score row (S = 2048, 256 threads x 8)
// In-place f32 normalize (d_out w-region) + bf16 copy for w@v GEMM.
// ---------------------------------------------------------------------------
__global__ __launch_bounds__(256) void softmax_row(float* __restrict__ sc,
                                                   const int* __restrict__ mask,
                                                   __bf16* __restrict__ wb,
                                                   int S_, float scale) {
    const int  i = blockIdx.x, b = blockIdx.y, tid = threadIdx.x;
    const long rowoff = ((long)b * S_ + i) * (long)S_;
    float*      r = sc + rowoff;
    __bf16*     o = wb + rowoff;
    const int* mr = mask + (long)b * S_;

    const int c0 = tid * 8;
    float v[8];
    float4 a0 = *(const float4*)&r[c0];
    float4 a1 = *(const float4*)&r[c0 + 4];
    v[0]=a0.x; v[1]=a0.y; v[2]=a0.z; v[3]=a0.w;
    v[4]=a1.x; v[5]=a1.y; v[6]=a1.z; v[7]=a1.w;
    #pragma unroll
    for (int j = 0; j < 8; j++) {
        v[j] *= scale;
        if (mr[c0 + j] == 0) v[j] = -1e9f;
    }

    float mx = v[0];
    #pragma unroll
    for (int j = 1; j < 8; j++) mx = fmaxf(mx, v[j]);
    mx = wave_max(mx);
    __shared__ float red[8];
    if ((tid & 31) == 0) red[tid >> 5] = mx;
    __syncthreads();
    float bm = red[0];
    #pragma unroll
    for (int j = 1; j < 8; j++) bm = fmaxf(bm, red[j]);
    __syncthreads();

    float sum = 0.f;
    #pragma unroll
    for (int j = 0; j < 8; j++) { v[j] = __expf(v[j] - bm); sum += v[j]; }
    sum = wave_sum(sum);
    if ((tid & 31) == 0) red[tid >> 5] = sum;
    __syncthreads();
    float bs = 0.f;
    #pragma unroll
    for (int j = 0; j < 8; j++) bs += red[j];
    const float inv = 1.f / bs;

    float4 w0, w1;
    w0.x=v[0]*inv; w0.y=v[1]*inv; w0.z=v[2]*inv; w0.w=v[3]*inv;
    w1.x=v[4]*inv; w1.y=v[5]*inv; w1.z=v[6]*inv; w1.w=v[7]*inv;
    *(float4*)&r[c0]     = w0;
    *(float4*)&r[c0 + 4] = w1;
    #pragma unroll
    for (int j = 0; j < 8; j++) o[c0 + j] = (__bf16)(v[j] * inv);
}

// ---------------------------------------------------------------------------
// bf16 GEMM, f32 accumulate via v_wmma_f32_16x16x32_bf16.
// C(MxN) = A(MxK) @ Bt^T, where Bt is stored N x K row-major (ldb = K stride).
// Block tile 128x128, 256 threads = 8 waves (2x4), wave tile 64x32 (4x2 frags).
// LDS double-buffered, K loop unrolled by 2 (REQUIRES K % 64 == 0), staging via
// CDNA5 GLOBAL_LOAD_ASYNC_TO_LDS_B128 (ASYNCcnt) -- no VGPR round trip.
// EPI: 0 = bf16 (+bias), 1 = f32 raw, 2 = f32 +bias+residual, 3 = bf16 relu(+bias)
// ---------------------------------------------------------------------------
template <int EPI>
__global__ __launch_bounds__(256)
void gemm_bf16(const __bf16* __restrict__ A, long aBatch, int lda,
               const __bf16* __restrict__ Bt, long bBatch, int ldb,
               const float* __restrict__ bias,
               void* __restrict__ Cout, long cBatch, int ldc,
               const float* __restrict__ Res, long rBatch,
               int M, int N, int K) {
    __shared__ __bf16 lsA[2][128 * 32];
    __shared__ __bf16 lsB[2][128 * 32];

    const int z = blockIdx.z;
    A  += (long)z * aBatch;
    Bt += (long)z * bBatch;

    const int tid  = threadIdx.x;
    const int m0   = blockIdx.y * 128, n0 = blockIdx.x * 128;
    const int wave = tid >> 5, lane = tid & 31;
    const int half = lane >> 4, lr = lane & 15;
    const int wm   = (wave >> 2) * 64, wn = (wave & 3) * 32;

    // --- staging: two 16B chunks per thread per operand, pointers hoisted ---
    const int c0 = tid * 2, c1 = c0 + 1;
    const int r0 = c0 >> 2, o0 = (c0 & 3) * 8;
    const int r1 = c1 >> 2, o1 = (c1 & 3) * 8;
    const __bf16* gA0 = A  + (long)(m0 + r0) * lda + o0;
    const __bf16* gA1 = A  + (long)(m0 + r1) * lda + o1;
    const __bf16* gB0 = Bt + (long)(n0 + r0) * ldb + o0;
    const __bf16* gB1 = Bt + (long)(n0 + r1) * ldb + o1;
    const int sA0 = r0 * 32 + o0, sA1 = r1 * 32 + o1;

    auto stage = [&](int buf) {
        __builtin_prefetch(gA0 + 64, 0, 3);
        __builtin_prefetch(gB0 + 64, 0, 3);
        async_ld_b128(gA0, &lsA[buf][sA0]);  gA0 += 32;
        async_ld_b128(gA1, &lsA[buf][sA1]);  gA1 += 32;
        async_ld_b128(gB0, &lsB[buf][sA0]);  gB0 += 32;
        async_ld_b128(gB1, &lsB[buf][sA1]);  gB1 += 32;
    };

    union AF { v16bf v; uint4 q[2]; };
    v8f acc[4][2];
    const v8f zero8 = {0.f, 0.f, 0.f, 0.f, 0.f, 0.f, 0.f, 0.f};
    #pragma unroll
    for (int i = 0; i < 4; i++)
        #pragma unroll
        for (int j = 0; j < 2; j++) acc[i][j] = zero8;

    auto compute = [&](int buf) {
        const __bf16* bA = lsA[buf];
        const __bf16* bB = lsB[buf];
        AF afr[4], bfr[2];
        #pragma unroll
        for (int fm = 0; fm < 4; fm++) {
            const uint4* p = (const uint4*)&bA[(wm + fm * 16 + lr) * 32];
            afr[fm].q[0] = p[half];
            afr[fm].q[1] = p[2 + half];
        }
        #pragma unroll
        for (int fn = 0; fn < 2; fn++) {
            const uint4* p = (const uint4*)&bB[(wn + fn * 16 + lr) * 32];
            bfr[fn].q[0] = p[half];
            bfr[fn].q[1] = p[2 + half];
        }
        #pragma unroll
        for (int fm = 0; fm < 4; fm++)
            #pragma unroll
            for (int fn = 0; fn < 2; fn++)
                acc[fm][fn] = __builtin_amdgcn_wmma_f32_16x16x32_bf16(
                    false, afr[fm].v, false, bfr[fn].v,
                    (short)0, acc[fm][fn], false, false);
    };

    // K % 64 == 0 required: unroll-by-2 keeps buffer indices literal.
    stage(0);
    wait_async();
    __syncthreads();
    for (int k0 = 0; k0 < K; k0 += 64) {
        stage(1);            // always valid: k0 + 32 < K inside the loop
        compute(0);
        wait_async();        // buf1 DMA complete before the barrier releases it
        __syncthreads();
        if (k0 + 64 < K) stage(0);
        compute(1);
        wait_async();
        __syncthreads();
    }

    // --- epilogue ---
    const long cb = (long)z * cBatch;
    float bcol[2] = {0.f, 0.f};
    if (EPI != 1) {
        #pragma unroll
        for (int fn = 0; fn < 2; fn++)
            bcol[fn] = bias ? bias[n0 + wn + fn * 16 + lr] : 0.f;
    }
    #pragma unroll
    for (int fm = 0; fm < 4; fm++) {
        #pragma unroll
        for (int fn = 0; fn < 2; fn++) {
            const int col = n0 + wn + fn * 16 + lr;
            #pragma unroll
            for (int r = 0; r < 8; r++) {
                const int row = m0 + wm + fm * 16 + r + 8 * half;
                float v = acc[fm][fn][r];
                const long off = cb + (long)row * ldc + col;
                if (EPI == 0) {
                    ((__bf16*)Cout)[off] = (__bf16)(v + bcol[fn]);
                } else if (EPI == 1) {
                    ((float*)Cout)[off] = v;
                } else if (EPI == 2) {
                    v += bcol[fn] + Res[(long)z * rBatch + (long)row * ldc + col];
                    ((float*)Cout)[off] = v;
                } else {
                    v += bcol[fn];
                    ((__bf16*)Cout)[off] = (__bf16)(v > 0.f ? v : 0.f);
                }
            }
        }
    }
}

// ---------------------------------------------------------------------------
// Host launcher
// ---------------------------------------------------------------------------
extern "C" void kernel_launch(void* const* d_in, const int* in_sizes, int n_in,
                              void* d_out, int out_size, void* d_ws, size_t ws_size,
                              hipStream_t stream) {
    constexpr int  Bn = 8, S = 2048, D = 1024, DFF = 1024;
    const long BS = (long)Bn * S;

    const float* x    = (const float*)d_in[0];
    const int*   mask = (const int*)d_in[1];
    const float* ln1g = (const float*)d_in[2];
    const float* ln1b = (const float*)d_in[3];
    const float* Wq   = (const float*)d_in[4];
    const float* bq   = (const float*)d_in[5];
    const float* Wk   = (const float*)d_in[6];
    const float* bk   = (const float*)d_in[7];
    const float* Wo   = (const float*)d_in[8];
    const float* bo   = (const float*)d_in[9];
    const float* ln2g = (const float*)d_in[10];
    const float* ln2b = (const float*)d_in[11];
    const float* W1   = (const float*)d_in[12];
    const float* b1   = (const float*)d_in[13];
    const float* W2   = (const float*)d_in[14];
    const float* b2   = (const float*)d_in[15];

    float* outp = (float*)d_out;
    float* wp   = outp + BS * D;   // f32 attention weights region of d_out

    char* ws = (char*)d_ws;
    const size_t MB = 1ull << 20;
    __bf16* XN   = (__bf16*)(ws);                 // BS*D bf16 : 32 MiB
    __bf16* Qb   = (__bf16*)(ws + 32 * MB);       // 32 MiB
    __bf16* Kb   = (__bf16*)(ws + 64 * MB);       // 32 MiB (v == k), S x D per batch
    __bf16* ATT  = (__bf16*)(ws + 96 * MB);       // 32 MiB
    float*  X2   = (float*) (ws + 128 * MB);      // BS*D f32 : 64 MiB
    __bf16* WB   = (__bf16*)(ws + 192 * MB);      // BS*S bf16 : 64 MiB
    __bf16* XN2  = XN;                            // reuse (xn dead after q,k)
    __bf16* H    = WB;                            // reuse (w_bf dead after attn)
    __bf16* KbT  = (__bf16*)(ws + 256 * MB);      // D x S per batch : 32 MiB
    __bf16* WqT  = (__bf16*)(ws + 288 * MB);      // weights, N x K bf16
    __bf16* WkT  = WqT + (size_t)D * D;
    __bf16* WoT  = WkT + (size_t)D * D;
    __bf16* W1T  = WoT + (size_t)D * D;           // DFF x D
    __bf16* W2T  = W1T + (size_t)D * DFF;         // D x DFF

    const dim3 blk(256);

    // weights: f32 (K x N) -> bf16 transposed (N x K)
    transpose_to_bf16<float><<<dim3(D / 32, D / 32, 1), blk, 0, stream>>>(Wq, WqT, D, D, 0, 0);
    transpose_to_bf16<float><<<dim3(D / 32, D / 32, 1), blk, 0, stream>>>(Wk, WkT, D, D, 0, 0);
    transpose_to_bf16<float><<<dim3(D / 32, D / 32, 1), blk, 0, stream>>>(Wo, WoT, D, D, 0, 0);
    transpose_to_bf16<float><<<dim3(DFF / 32, D / 32, 1), blk, 0, stream>>>(W1, W1T, D, DFF, 0, 0);
    transpose_to_bf16<float><<<dim3(D / 32, DFF / 32, 1), blk, 0, stream>>>(W2, W2T, DFF, D, 0, 0);

    // LN1
    ln_bf16<<<dim3((unsigned)BS), blk, 0, stream>>>(x, ln1g, ln1b, XN, D);

    // q = xn@Wq + bq ; k = v = xn@Wk + bk   (bf16 out)
    gemm_bf16<0><<<dim3(D / 128, BS / 128, 1), blk, 0, stream>>>(
        XN, 0, D, WqT, 0, D, bq, Qb, 0, D, nullptr, 0, (int)BS, D, D);
    gemm_bf16<0><<<dim3(D / 128, BS / 128, 1), blk, 0, stream>>>(
        XN, 0, D, WkT, 0, D, bk, Kb, 0, D, nullptr, 0, (int)BS, D, D);

    // per-batch v^T = k^T (D x S) for the w@v GEMM's fast staging path
    transpose_to_bf16<__bf16><<<dim3(D / 32, S / 32, Bn), blk, 0, stream>>>(
        Kb, KbT, S, D, (long)S * D, (long)S * D);

    // scores = q @ k^T  (k is S x D = N x K row-major already; raw f32 into d_out)
    gemm_bf16<1><<<dim3(S / 128, S / 128, Bn), blk, 0, stream>>>(
        Qb, (long)S * D, D, Kb, (long)S * D, D, nullptr,
        wp, (long)S * S, S, nullptr, 0, S, S, D);

    // masked softmax (scale 1/sqrt(D)); normalize in place + bf16 copy
    softmax_row<<<dim3(S, Bn), blk, 0, stream>>>(wp, mask, WB, S, 0.03125f);

    // attn = w @ v   (Bt = v^T, D x S)
    gemm_bf16<0><<<dim3(D / 128, S / 128, Bn), blk, 0, stream>>>(
        WB, (long)S * S, S, KbT, (long)S * D, S, nullptr,
        ATT, (long)S * D, D, nullptr, 0, S, D, S);

    // x2 = x + attn@Wo + bo   (f32)
    gemm_bf16<2><<<dim3(D / 128, BS / 128, 1), blk, 0, stream>>>(
        ATT, 0, D, WoT, 0, D, bo, X2, 0, D, x, 0, (int)BS, D, D);

    // LN2
    ln_bf16<<<dim3((unsigned)BS), blk, 0, stream>>>(X2, ln2g, ln2b, XN2, D);

    // h = relu(xn2@W1 + b1)
    gemm_bf16<3><<<dim3(DFF / 128, BS / 128, 1), blk, 0, stream>>>(
        XN2, 0, D, W1T, 0, DFF, b1, H, 0, DFF, nullptr, 0, (int)BS, DFF, D);

    // out = x2 + h@W2 + b2
    gemm_bf16<2><<<dim3(D / 128, BS / 128, 1), blk, 0, stream>>>(
        H, 0, DFF, W2T, 0, DFF, b2, outp, 0, D, X2, 0, (int)BS, D, DFF);
}
